// NeuralNet_13365938225728
// MI455X (gfx1250) — matrix operand order repaired and checked by reading, not compiled, not run on hardware
//
#include <hip/hip_runtime.h>

typedef __attribute__((ext_vector_type(16))) _Float16 v16h;
typedef __attribute__((ext_vector_type(8)))  _Float16 v8h;
typedef __attribute__((ext_vector_type(8)))  float    v8f;
typedef __attribute__((ext_vector_type(4)))  float    v4f;

#define NB   4096   // batch
#define DD   768    // embedding dim
#define FF   2304   // 3*D
#define KT_D (DD / 32)   // 24 k-tiles per segment
#define BN_EPS 1e-3f

// ============================================================================
// Weight prepack: f32 row-major W[K][ld] -> f16 WMMA B-fragment layout.
// Fragment order (per 32x16 tile): ((kt*NT + t)*32 + lane)*16 + e, where
// lane<16 holds K=kt*32+{0..7,16..23}, lane>=16 holds K=kt*32+{8..15,24..31}, N = t*16 + lane%16.
// Columns >= ld are zero-padded.
// ============================================================================
__global__ void prepack_kernel(const float* __restrict__ W, int ld, int NT,
                               int total, _Float16* __restrict__ out) {
  int idx = blockIdx.x * blockDim.x + threadIdx.x;
  if (idx >= total) return;
  int e    = idx & 15;
  int lane = (idx >> 4) & 31;
  int tile = idx >> 9;
  int t    = tile % NT;
  int kt   = tile / NT;
  int k = kt * 32 + ((lane < 16) ? 0 : 8) + e + ((e < 8) ? 0 : 8);
  int n = t * 16 + (lane & 15);
  out[idx] = (n < ld) ? (_Float16)W[(size_t)k * ld + n] : (_Float16)0.f;
}

// ---- fragment loaders ------------------------------------------------------
// A-matrix 16x32 f16: lane%16 = M; lanes<16 hold K {0..7,16..23}, lanes>=16
// hold K {8..15,24..31}. Two 32B contiguous f32 runs -> 4x b128 loads.
__device__ __forceinline__ v16h load_a_g(const float* __restrict__ seg,
                                         int off, float s, int lane) {
  const float* p = seg + off + ((lane < 16) ? 0 : 8);
  v4f x0 = *(const v4f*)(p + 0);
  v4f x1 = *(const v4f*)(p + 4);
  v4f y0 = *(const v4f*)(p + 16);
  v4f y1 = *(const v4f*)(p + 20);
  v16h a;
#pragma unroll
  for (int i = 0; i < 4; ++i) {
    a[i]      = (_Float16)(s * x0[i]);
    a[4 + i]  = (_Float16)(s * x1[i]);
    a[8 + i]  = (_Float16)(s * y0[i]);
    a[12 + i] = (_Float16)(s * y1[i]);
  }
  return a;
}

// A fragment from an f16 LDS row (contiguous 16B runs -> 2x ds_load_b128)
__device__ __forceinline__ v16h load_a_lds(const _Float16* __restrict__ row,
                                           int k0, int lane) {
  const int base = k0 + ((lane < 16) ? 0 : 8);
  v8h lo = *(const v8h*)(row + base);
  v8h hi = *(const v8h*)(row + base + 16);
  return __builtin_shufflevector(lo, hi, 0, 1, 2, 3, 4, 5, 6, 7,
                                 8, 9, 10, 11, 12, 13, 14, 15);
}

// Prepacked B fragment: one 32B contiguous load per lane
__device__ __forceinline__ v16h load_bp(const v16h* __restrict__ Wp,
                                        int kt, int NT, int t, int lane) {
  return Wp[((size_t)(kt * NT + t)) * 32 + lane];
}

// ---- kernel 1: attention scalar w per row ---------------------------------
__global__ __launch_bounds__(128) void attn_kernel(
    const float* __restrict__ q, const float* __restrict__ nodes,
    const float* __restrict__ leafs, const int* __restrict__ positives,
    const int* __restrict__ negatives, const int* __restrict__ p_isleafs,
    const int* __restrict__ n_isleafs, const v16h* __restrict__ W00p,
    const float* __restrict__ b00, const float* __restrict__ a00,
    const float* __restrict__ W01, const float* __restrict__ b01,
    float* __restrict__ w_out) {
  const int lane = threadIdx.x & 31;
  const int wave = threadIdx.x >> 5;
  const int m0   = (blockIdx.x * (blockDim.x >> 5) + wave) * 16;
  const int m    = m0 + (lane & 15);

  const float* qrow = q + (size_t)m * DD;
  const float* prow = ((p_isleafs[m] == 1) ? leafs : nodes) + (size_t)positives[m] * DD;
  const float* nrow = ((n_isleafs[m] == 1) ? leafs : nodes) + (size_t)negatives[m] * DD;

  v8f c[3] = {};
  // feat = [q | pos | neg]; straight-line loop per segment (no per-iter select)
  auto seg_gemm = [&](const float* __restrict__ seg, int ktoff) {
    for (int kk = 0; kk < DD; kk += 32) {
      __builtin_prefetch(seg + kk + 64, 0, 1);  // global_prefetch_b8
      v16h a = load_a_g(seg, kk, 1.0f, lane);
      const int kt = ktoff + (kk >> 5);
#pragma unroll
      for (int t = 0; t < 3; ++t) {
        v16h b = load_bp(W00p, kt, 3, t, lane);
        c[t] = __builtin_amdgcn_wmma_f32_16x16x32_f16(false, a, false, b,
                                                      (short)0, c[t], false, false);
      }
    }
  };
  seg_gemm(qrow, 0);
  seg_gemm(prow, KT_D);
  seg_gemm(nrow, 2 * KT_D);

  // epilogue: h = prelu(c + b00), partial w = h . W01; reduce over N lanes
  float part[8];
#pragma unroll
  for (int r = 0; r < 8; ++r) part[r] = 0.f;
#pragma unroll
  for (int t = 0; t < 3; ++t) {
    int n = t * 16 + (lane & 15);
    if (n < 36) {
      float bb = b00[n], aa = a00[n], ww = W01[n];
#pragma unroll
      for (int r = 0; r < 8; ++r) {
        float h = c[t][r] + bb;
        h = fmaxf(h, 0.f) + aa * fminf(h, 0.f);
        part[r] += h * ww;
      }
    }
  }
  // reduce across the 16 lanes of each half-wave (M=r in lanes<16, M=r+8 above)
#pragma unroll
  for (int r = 0; r < 8; ++r) {
#pragma unroll
    for (int off = 1; off < 16; off <<= 1)
      part[r] += __shfl_xor(part[r], off, 32);
  }
  if ((lane & 15) == 0) {
    float bb = b01[0];
    int base = m0 + ((lane >> 4) ? 8 : 0);
#pragma unroll
    for (int r = 0; r < 8; ++r) w_out[base + r] = part[r] + bb;
  }
}

// ---- kernel 2: MLP 2304->128->64->24->2 + softmax -------------------------
__global__ __launch_bounds__(256) void mlp_kernel(
    const float* __restrict__ q, const float* __restrict__ nodes,
    const float* __restrict__ leafs, const int* __restrict__ positives,
    const int* __restrict__ negatives, const int* __restrict__ p_isleafs,
    const int* __restrict__ n_isleafs, const float* __restrict__ w_in,
    const v16h* __restrict__ W1p, const float* __restrict__ b1,
    const float* __restrict__ a1, const float* __restrict__ g1,
    const float* __restrict__ be1, const float* __restrict__ m1,
    const float* __restrict__ v1, const v16h* __restrict__ W2p,
    const float* __restrict__ b2, const float* __restrict__ a2,
    const float* __restrict__ g2, const float* __restrict__ be2,
    const float* __restrict__ m2, const float* __restrict__ v2,
    const v16h* __restrict__ W3p, const float* __restrict__ b3,
    const float* __restrict__ a3, const float* __restrict__ g3,
    const float* __restrict__ be3, const float* __restrict__ m3,
    const float* __restrict__ v3, const float* __restrict__ W4,
    const float* __restrict__ b4, float* __restrict__ out) {
  __shared__ _Float16 xbuf[8][16][128];  // per-wave activation tile (f16)
  __shared__ float    fbuf[8][16][24];   // per-wave layer-3 output (f32)

  const int lane = threadIdx.x & 31;
  const int wave = threadIdx.x >> 5;
  const int m0   = (blockIdx.x * (blockDim.x >> 5) + wave) * 16;
  const int m    = m0 + (lane & 15);

  const float* qrow = q + (size_t)m * DD;
  const float* prow = ((p_isleafs[m] == 1) ? leafs : nodes) + (size_t)positives[m] * DD;
  const float* nrow = ((n_isleafs[m] == 1) ? leafs : nodes) + (size_t)negatives[m] * DD;
  const float  wm   = w_in[m];

  const int mloc = (lane < 16) ? 0 : 8;  // C fragment M base for this half
  const int n16  = lane & 15;            // C fragment N for this lane

  // ---- layer 1: item = [w*pos | w*neg | q] @ W1[2304x128] ----
  v8f c1[8] = {};
  auto seg_gemm1 = [&](const float* __restrict__ seg, float s, int ktoff) {
    for (int kk = 0; kk < DD; kk += 32) {
      __builtin_prefetch(seg + kk + 64, 0, 1);  // global_prefetch_b8
      v16h a = load_a_g(seg, kk, s, lane);
      const int kt = ktoff + (kk >> 5);
#pragma unroll
      for (int t = 0; t < 8; ++t) {
        v16h b = load_bp(W1p, kt, 8, t, lane);
        c1[t] = __builtin_amdgcn_wmma_f32_16x16x32_f16(false, a, false, b,
                                                       (short)0, c1[t], false, false);
      }
    }
  };
  seg_gemm1(prow, wm, 0);
  seg_gemm1(nrow, wm, KT_D);
  seg_gemm1(qrow, 1.0f, 2 * KT_D);

  // bn(prelu()) -> stage as f16 in LDS
#pragma unroll
  for (int t = 0; t < 8; ++t) {
    int nn = t * 16 + n16;
    float bb = b1[nn], aa = a1[nn];
    float inv = rsqrtf(v1[nn] + BN_EPS) * g1[nn];
    float mm = m1[nn], bee = be1[nn];
#pragma unroll
    for (int r = 0; r < 8; ++r) {
      float h = c1[t][r] + bb;
      h = fmaxf(h, 0.f) + aa * fminf(h, 0.f);
      h = (h - mm) * inv + bee;
      xbuf[wave][mloc + r][nn] = (_Float16)h;
    }
  }
  __syncthreads();

  // ---- layer 2: x1[16x128] @ W2[128x64] ----
  v8f c2[4] = {};
#pragma unroll
  for (int k0 = 0; k0 < 128; k0 += 32) {
    v16h a = load_a_lds(&xbuf[wave][lane & 15][0], k0, lane);
    const int kt = k0 >> 5;
#pragma unroll
    for (int t = 0; t < 4; ++t) {
      v16h b = load_bp(W2p, kt, 4, t, lane);
      c2[t] = __builtin_amdgcn_wmma_f32_16x16x32_f16(false, a, false, b,
                                                     (short)0, c2[t], false, false);
    }
  }
  __syncthreads();
#pragma unroll
  for (int t = 0; t < 4; ++t) {
    int nn = t * 16 + n16;
    float bb = b2[nn], aa = a2[nn];
    float inv = rsqrtf(v2[nn] + BN_EPS) * g2[nn];
    float mm = m2[nn], bee = be2[nn];
#pragma unroll
    for (int r = 0; r < 8; ++r) {
      float h = c2[t][r] + bb;
      h = fmaxf(h, 0.f) + aa * fminf(h, 0.f);
      h = (h - mm) * inv + bee;
      xbuf[wave][mloc + r][nn] = (_Float16)h;
    }
  }
  __syncthreads();

  // ---- layer 3: x2[16x64] @ W3[64x24] ----
  v8f c3[2] = {};
#pragma unroll
  for (int k0 = 0; k0 < 64; k0 += 32) {
    v16h a = load_a_lds(&xbuf[wave][lane & 15][0], k0, lane);
    const int kt = k0 >> 5;
#pragma unroll
    for (int t = 0; t < 2; ++t) {
      v16h b = load_bp(W3p, kt, 2, t, lane);
      c3[t] = __builtin_amdgcn_wmma_f32_16x16x32_f16(false, a, false, b,
                                                     (short)0, c3[t], false, false);
    }
  }
  __syncthreads();
#pragma unroll
  for (int t = 0; t < 2; ++t) {
    int nn = t * 16 + n16;
    if (nn < 24) {
      float bb = b3[nn], aa = a3[nn];
      float inv = rsqrtf(v3[nn] + BN_EPS) * g3[nn];
      float mm = m3[nn], bee = be3[nn];
#pragma unroll
      for (int r = 0; r < 8; ++r) {
        float h = c3[t][r] + bb;
        h = fmaxf(h, 0.f) + aa * fminf(h, 0.f);
        h = (h - mm) * inv + bee;
        fbuf[wave][mloc + r][nn] = h;
      }
    }
  }
  __syncthreads();

  // ---- layer 4: x3[16x24] @ W4[24x2] + softmax (scalar, lanes 0..15) ----
  if (lane < 16) {
    int row = m0 + lane;
    float z0 = b4[0], z1 = b4[1];
#pragma unroll
    for (int j = 0; j < 24; ++j) {
      float x = fbuf[wave][lane][j];
      z0 += x * W4[j * 2 + 0];
      z1 += x * W4[j * 2 + 1];
    }
    float mz = fmaxf(z0, z1);
    float e0 = expf(z0 - mz), e1 = expf(z1 - mz);
    float inv = 1.f / (e0 + e1);
    out[(size_t)row * 2 + 0] = e0 * inv;
    out[(size_t)row * 2 + 1] = e1 * inv;
  }
}

extern "C" void kernel_launch(void* const* d_in, const int* in_sizes, int n_in,
                              void* d_out, int out_size, void* d_ws, size_t ws_size,
                              hipStream_t stream) {
  const float* q     = (const float*)d_in[0];
  const float* nodes = (const float*)d_in[1];
  const float* leafs = (const float*)d_in[2];
  const int*   pos   = (const int*)d_in[3];
  const int*   neg   = (const int*)d_in[4];
  const int*   pis   = (const int*)d_in[5];
  const int*   nis   = (const int*)d_in[6];
  const float* W00 = (const float*)d_in[7];
  const float* b00 = (const float*)d_in[8];
  const float* a00 = (const float*)d_in[9];
  const float* W01 = (const float*)d_in[10];
  const float* b01 = (const float*)d_in[11];
  const float* W1 = (const float*)d_in[12], *b1 = (const float*)d_in[13],
             *a1 = (const float*)d_in[14], *g1 = (const float*)d_in[15],
             *be1 = (const float*)d_in[16], *m1 = (const float*)d_in[17],
             *v1 = (const float*)d_in[18];
  const float* W2 = (const float*)d_in[19], *b2 = (const float*)d_in[20],
             *a2 = (const float*)d_in[21], *g2 = (const float*)d_in[22],
             *be2 = (const float*)d_in[23], *m2 = (const float*)d_in[24],
             *v2 = (const float*)d_in[25];
  const float* W3 = (const float*)d_in[26], *b3 = (const float*)d_in[27],
             *a3 = (const float*)d_in[28], *g3 = (const float*)d_in[29],
             *be3 = (const float*)d_in[30], *m3 = (const float*)d_in[31],
             *v3 = (const float*)d_in[32];
  const float* W4 = (const float*)d_in[33], *b4 = (const float*)d_in[34];

  // workspace layout (all 32B aligned)
  char* ws = (char*)d_ws;
  float*    w_ws = (float*)ws;                      // NB floats
  size_t o = (size_t)NB * sizeof(float);            // 16384
  const int n00 = (FF / 32) * 3 * 512;              // W00 packed halves
  const int n1  = (FF / 32) * 8 * 512;
  const int n2  = (128 / 32) * 4 * 512;
  const int n3  = (64 / 32) * 2 * 512;
  _Float16* W00p = (_Float16*)(ws + o); o += (size_t)n00 * 2;
  _Float16* W1p  = (_Float16*)(ws + o); o += (size_t)n1 * 2;
  _Float16* W2p  = (_Float16*)(ws + o); o += (size_t)n2 * 2;
  _Float16* W3p  = (_Float16*)(ws + o); o += (size_t)n3 * 2;

  // prepack weights into WMMA B-fragment layout (cheap, L2-resident)
  prepack_kernel<<<(n00 + 255) / 256, 256, 0, stream>>>(W00, 36, 3, n00, W00p);
  prepack_kernel<<<(n1 + 255) / 256, 256, 0, stream>>>(W1, 128, 8, n1, W1p);
  prepack_kernel<<<(n2 + 255) / 256, 256, 0, stream>>>(W2, 64, 4, n2, W2p);
  prepack_kernel<<<(n3 + 255) / 256, 256, 0, stream>>>(W3, 24, 2, n3, W3p);

  // 4 waves/block * 16 rows = 64 rows/block
  attn_kernel<<<NB / 64, 128, 0, stream>>>(q, nodes, leafs, pos, neg, pis, nis,
                                           (const v16h*)W00p, b00, a00, W01, b01,
                                           w_ws);
  // 8 waves/block * 16 rows = 128 rows/block
  mlp_kernel<<<NB / 128, 256, 0, stream>>>(q, nodes, leafs, pos, neg, pis, nis,
                                           w_ws, (const v16h*)W1p, b1, a1, g1,
                                           be1, m1, v1, (const v16h*)W2p, b2,
                                           a2, g2, be2, m2, v2,
                                           (const v16h*)W3p, b3, a3, g3, be3,
                                           m3, v3, W4, b4, (float*)d_out);
}
